// TransformerEncoderLayer_7713761263846
// MI455X (gfx1250) — compile-verified, matrix operand order checked by reading, not run on hardware
//
#include <hip/hip_runtime.h>
#include <math.h>

typedef __bf16 bf16;
typedef __attribute__((ext_vector_type(16))) __bf16 bf16x16;
typedef __attribute__((ext_vector_type(8)))  __bf16 bf16x8;
typedef __attribute__((ext_vector_type(8)))  float  f32x8;

#define DM   768
#define NH   12
#define HD   64
#define DFF  3072
#define SEQL 2048
#define BS   4
#define NTOK (SEQL*BS)   // 8192
#define NBH  (BS*NH)     // 48

// ---------------- WMMA helpers ----------------

__device__ __forceinline__ f32x8 wmma_bf(bf16x16 a, bf16x16 b, f32x8 c) {
  return __builtin_amdgcn_wmma_f32_16x16x32_bf16(false, a, false, b, (short)0, c,
                                                 false, false);
}

__device__ __forceinline__ f32x8 zero8() {
  f32x8 z = {0.f,0.f,0.f,0.f,0.f,0.f,0.f,0.f};
  return z;
}

// 16x32 bf16 fragment load (A-operand layout; also B-operand from Bt=NxK rows).
__device__ __forceinline__ bf16x16 load_frag(const bf16* __restrict__ base, int ld) {
  const int lane = threadIdx.x & 31;
  const int row  = lane & 15;
  const int colb = (lane >> 4) << 3;           // 0 or 8
  const bf16* p = base + (size_t)row * ld + colb;
  bf16x8 lo = *(const bf16x8*)(p);
  bf16x8 hi = *(const bf16x8*)(p + 16);
  bf16x16 f;
#pragma unroll
  for (int i = 0; i < 8; ++i) { f[i] = lo[i]; f[i+8] = hi[i]; }
  return f;
}

// ---------------- DPP 16-lane butterfly reductions (pure VALU) ----------------

template<int CTRL>
__device__ __forceinline__ float dpp_mov(float x) {
  int xi = __float_as_int(x);
  return __int_as_float(__builtin_amdgcn_update_dpp(xi, xi, CTRL, 0xF, 0xF, false));
}
// xor1 = quad_perm[1,0,3,2]=0xB1 ; xor2 = quad_perm[2,3,0,1]=0x4E
// lane^7 = ROW_HALF_MIRROR (0x141) ; lane^15 = ROW_MIRROR (0x140)
__device__ __forceinline__ float red16_max(float x) {
  x = fmaxf(x, dpp_mov<0xB1>(x));
  x = fmaxf(x, dpp_mov<0x4E>(x));
  x = fmaxf(x, dpp_mov<0x141>(x));
  x = fmaxf(x, dpp_mov<0x140>(x));
  return x;
}
__device__ __forceinline__ float red16_sum(float x) {
  x += dpp_mov<0xB1>(x);
  x += dpp_mov<0x4E>(x);
  x += dpp_mov<0x141>(x);
  x += dpp_mov<0x140>(x);
  return x;
}

// ---------------- CDNA5 async copy (global -> LDS, ASYNCcnt) ----------------

__device__ __forceinline__ void async_b128(const bf16* gptr, bf16* lptr) {
  unsigned la = (unsigned)(uintptr_t)lptr;         // low 32 bits = LDS offset
  unsigned long long ga = (unsigned long long)(uintptr_t)gptr;
  asm volatile("global_load_async_to_lds_b128 %0, %1, off"
               :: "v"(la), "v"(ga) : "memory");
}
__device__ __forceinline__ void wait_async0() {
  asm volatile("s_wait_asynccnt 0x0" ::: "memory");
}
template<int N>
__device__ __forceinline__ void wait_asyncN() {
  asm volatile("s_wait_asynccnt %0" :: "i"(N) : "memory");
}

// ---------------- prep kernels ----------------

__global__ __launch_bounds__(256) void cvt_kernel(const float* __restrict__ in,
                                                  bf16* __restrict__ out, int n) {
  int i = blockIdx.x * 256 + threadIdx.x;
  if (i < n) out[i] = (bf16)in[i];
}

// out[n*K + k] = (bf16) in[k*N + n]
__global__ __launch_bounds__(256) void tr_kernel(const float* __restrict__ in,
                                                 bf16* __restrict__ out, int K, int N) {
  int i = blockIdx.x * 256 + threadIdx.x;
  if (i < K * N) {
    int k = i / N, n = i - k * N;
    out[(size_t)n * K + k] = (bf16)in[i];
  }
}

// ---------------- GEMM: out = A(MxK) * Bt(NxK)^T + bias --------------------
// 8 waves (4M x 2N), wave tile 32x64, block tile 128x128.
// K-slices (128x64 of A and Bt) staged in LDS via async loads, double-buffered:
// 16 WMMAs per wave per sync epoch, 8 in-flight async b128 per thread.

enum { EPI_QK = 0, EPI_V = 1, EPI_GELU = 2, EPI_F32 = 3 };

template<int KDIM, int NDIM, int EPI>
__global__ __launch_bounds__(256) void gemm_kernel(
    const bf16* __restrict__ A, const bf16* __restrict__ Bt,
    const float* __restrict__ bias, void* __restrict__ outv)
{
  __shared__ __align__(16) bf16 sA[2][128 * 64];
  __shared__ __align__(16) bf16 sB[2][128 * 64];

  const int tid   = threadIdx.x;
  const int w     = tid >> 5;
  const int lane  = tid & 31;
  const int mBlk  = blockIdx.x * 128;
  const int nBlk  = blockIdx.y * 128;
  const int mW    = (w & 3) * 32;
  const int nW    = (w >> 2) * 64;

  // 128x64 tile = 1024 x 16B chunks: chunk c -> row=c>>3, kcol=(c&7)*8
  auto issue = [&](int buf, int k0) {
#pragma unroll
    for (int t = 0; t < 4; ++t) {
      const int c = tid + t * 256;
      const int row = c >> 3, kc = (c & 7) * 8;
      async_b128(A + (size_t)(mBlk + row) * KDIM + k0 + kc, &sA[buf][row * 64 + kc]);
    }
#pragma unroll
    for (int t = 0; t < 4; ++t) {
      const int c = tid + t * 256;
      const int row = c >> 3, kc = (c & 7) * 8;
      async_b128(Bt + (size_t)(nBlk + row) * KDIM + k0 + kc, &sB[buf][row * 64 + kc]);
    }
  };

  f32x8 acc[2][4];
#pragma unroll
  for (int i = 0; i < 2; ++i)
#pragma unroll
    for (int j = 0; j < 4; ++j) acc[i][j] = zero8();

  issue(0, 0);
  for (int k = 0, it = 0; k < KDIM; k += 64, ++it) {
    const int cur = it & 1;
    if (k + 64 < KDIM) { issue(cur ^ 1, k + 64); wait_asyncN<8>(); }
    else               { wait_async0(); }
    __syncthreads();

#pragma unroll
    for (int kk = 0; kk < 64; kk += 32) {
      bf16x16 af[2];
#pragma unroll
      for (int i = 0; i < 2; ++i)
        af[i] = load_frag(&sA[cur][(mW + 16 * i) * 64 + kk], 64);
#pragma unroll
      for (int j = 0; j < 4; ++j) {
        bf16x16 bfr = load_frag(&sB[cur][(nW + 16 * j) * 64 + kk], 64);
#pragma unroll
        for (int i = 0; i < 2; ++i) acc[i][j] = wmma_bf(af[i], bfr, acc[i][j]);
      }
    }
    __syncthreads();
  }

  const int rOff = (lane >> 4) * 8;
  const int col  = lane & 15;
#pragma unroll
  for (int i = 0; i < 2; ++i)
#pragma unroll
    for (int j = 0; j < 4; ++j)
#pragma unroll
      for (int r = 0; r < 8; ++r) {
        const int m = mBlk + mW + 16 * i + r + rOff;
        const int n = nBlk + nW + 16 * j + col;
        float v = acc[i][j][r] + bias[n];
        if (EPI == EPI_F32) {
          ((float*)outv)[(size_t)m * NDIM + n] = v;
        } else if (EPI == EPI_GELU) {
          v = 0.5f * v * (1.0f + erff(v * 0.70710678118654752f));
          ((bf16*)outv)[(size_t)m * NDIM + n] = (bf16)v;
        } else {
          const int l = m >> 2, b = m & 3;
          const int h = n >> 6, d = n & 63;
          if (EPI == EPI_QK)
            ((bf16*)outv)[(((size_t)(b*NH + h)) * SEQL + l) * HD + d] = (bf16)v;
          else
            ((bf16*)outv)[(((size_t)(b*NH + h)) * HD + d) * SEQL + l] = (bf16)v;
        }
      }
}

// ---------------- flash attention (no 1/sqrt(hd) scaling) ------------------
// grid (SEQL/128, NBH); 8 waves/block, one wave = 16 query rows; 32 keys/iter.
// K (32x64) and Vt (64x32) tiles async-staged in LDS, shared by all 8 waves;
// softmax reductions are DPP-only VALU (co-executes with XDL WMMAs).

__global__ __launch_bounds__(256) void attn_kernel(
    const bf16* __restrict__ Q, const bf16* __restrict__ K,
    const bf16* __restrict__ Vt, bf16* __restrict__ AO)
{
  __shared__ __align__(16) bf16 sK[2][32 * 64];
  __shared__ __align__(16) bf16 sV[2][64 * 32];
  __shared__ __align__(16) bf16 pbuf[8][16 * 32];

  const int tid   = threadIdx.x;
  const int w     = tid >> 5;
  const int lane  = tid & 31;
  const int bh    = blockIdx.y;
  const int qBase = blockIdx.x * 128 + w * 16;

  const bf16* Qb = Q  + (size_t)bh * SEQL * HD;
  const bf16* Kb = K  + (size_t)bh * SEQL * HD;
  const bf16* Vb = Vt + (size_t)bh * HD * SEQL;

  const int kr = tid >> 3, kk = (tid & 7) * 8;   // K tile 32x64
  const int vr = tid >> 2, vk = (tid & 3) * 8;   // V tile 64x32

  auto issue = [&](int buf, int kb) {
    async_b128(Kb + (size_t)(kb + kr) * HD + kk, &sK[buf][kr * 64 + kk]);
    async_b128(Vb + (size_t)vr * SEQL + kb + vk, &sV[buf][vr * 32 + vk]);
  };

  const bf16x16 qf0 = load_frag(Qb + (size_t)qBase * HD + 0,  HD);
  const bf16x16 qf1 = load_frag(Qb + (size_t)qBase * HD + 32, HD);

  f32x8 o[4];
#pragma unroll
  for (int j = 0; j < 4; ++j) o[j] = zero8();
  float mrow[8], ssum[8];
#pragma unroll
  for (int r = 0; r < 8; ++r) { mrow[r] = -3.0e38f; ssum[r] = 0.f; }

  bf16* myp = &pbuf[w][0];
  const int rOff = (lane >> 4) * 8;
  const int col  = lane & 15;

  issue(0, 0);
  for (int kb = 0, it = 0; kb < SEQL; kb += 32, ++it) {
    const int cur = it & 1;
    if (kb + 32 < SEQL) { issue(cur ^ 1, kb + 32); wait_asyncN<2>(); }
    else                { wait_async0(); }
    __syncthreads();

    // S = Q * K^T (two 16-key tiles, K-dim 64 = 2 WMMA each)
    f32x8 s0 = zero8(), s1 = zero8();
    {
      bf16x16 kf;
      kf = load_frag(&sK[cur][ 0 * 64 + 0],  64); s0 = wmma_bf(qf0, kf, s0);
      kf = load_frag(&sK[cur][ 0 * 64 + 32], 64); s0 = wmma_bf(qf1, kf, s0);
      kf = load_frag(&sK[cur][16 * 64 + 0],  64); s1 = wmma_bf(qf0, kf, s1);
      kf = load_frag(&sK[cur][16 * 64 + 32], 64); s1 = wmma_bf(qf1, kf, s1);
    }

    // online softmax (per C-reg row; DPP butterfly inside 16-lane halves)
    f32x8 p0, p1;
#pragma unroll
    for (int r = 0; r < 8; ++r) {
      const float v  = red16_max(fmaxf(s0[r], s1[r]));
      const float mn = fmaxf(mrow[r], v);
      const float sc = __expf(mrow[r] - mn);
      mrow[r] = mn;
#pragma unroll
      for (int j = 0; j < 4; ++j) o[j][r] *= sc;
      p0[r] = __expf(s0[r] - mn);
      p1[r] = __expf(s1[r] - mn);
      ssum[r] = ssum[r] * sc + red16_sum(p0[r] + p1[r]);
    }

    // C-layout -> A-layout via per-wave LDS bounce
#pragma unroll
    for (int r = 0; r < 8; ++r) {
      myp[(r + rOff) * 32 + col]      = (bf16)p0[r];
      myp[(r + rOff) * 32 + 16 + col] = (bf16)p1[r];
    }
    asm volatile("s_wait_dscnt 0x0" ::: "memory");
    const bf16x16 pf = load_frag(myp, 32);

    // O += P * V
#pragma unroll
    for (int j = 0; j < 4; ++j) {
      bf16x16 vf = load_frag(&sV[cur][(j * 16) * 32], 32);
      o[j] = wmma_bf(pf, vf, o[j]);
    }
    __syncthreads();
  }

  const int b = bh / NH, h = bh % NH;
#pragma unroll
  for (int j = 0; j < 4; ++j)
#pragma unroll
    for (int r = 0; r < 8; ++r) {
      const int l = qBase + r + rOff;
      const int d = j * 16 + col;
      const float v = o[j][r] / ssum[r];
      AO[((size_t)l * BS + b) * DM + h * HD + d] = (bf16)v;
    }
}

// ---------------- residual + layernorm ----------------

template<bool WBF>
__global__ __launch_bounds__(256) void ln_kernel(
    const float* __restrict__ xa, const float* __restrict__ xb,
    const float* __restrict__ g,  const float* __restrict__ beta,
    float* __restrict__ out, bf16* __restrict__ outbf)
{
  const int row = blockIdx.x;
  const int t = threadIdx.x;
  const int w = t >> 5, lane = t & 31;
  __shared__ float sh1[8], sh2[8];
  const size_t off = (size_t)row * DM;

  float v[3];
#pragma unroll
  for (int k = 0; k < 3; ++k) v[k] = xa[off + t + 256*k] + xb[off + t + 256*k];

  float s = v[0] + v[1] + v[2];
#pragma unroll
  for (int o = 1; o < 32; o <<= 1) s += __shfl_xor(s, o, 32);
  if (lane == 0) sh1[w] = s;
  __syncthreads();
  float tot = 0.f;
#pragma unroll
  for (int i = 0; i < 8; ++i) tot += sh1[i];
  const float mu = tot * (1.0f / DM);

  float d0 = v[0]-mu, d1 = v[1]-mu, d2 = v[2]-mu;
  float q = d0*d0 + d1*d1 + d2*d2;
#pragma unroll
  for (int o = 1; o < 32; o <<= 1) q += __shfl_xor(q, o, 32);
  if (lane == 0) sh2[w] = q;
  __syncthreads();
  float tq = 0.f;
#pragma unroll
  for (int i = 0; i < 8; ++i) tq += sh2[i];
  const float inv = rsqrtf(tq * (1.0f / DM) + 1e-5f);

  float dd[3] = {d0, d1, d2};
#pragma unroll
  for (int k = 0; k < 3; ++k) {
    const int c = t + 256*k;
    const float y = dd[k] * inv * g[c] + beta[c];
    out[off + c] = y;
    if (WBF) outbf[off + c] = (bf16)y;
  }
}

// ---------------- driver ----------------

extern "C" void kernel_launch(void* const* d_in, const int* in_sizes, int n_in,
                              void* d_out, int out_size, void* d_ws, size_t ws_size,
                              hipStream_t stream)
{
  (void)in_sizes; (void)n_in; (void)out_size; (void)ws_size;
  const float* src = (const float*)d_in[0];
  const float* Wq  = (const float*)d_in[1];  const float* bq  = (const float*)d_in[2];
  const float* Wk  = (const float*)d_in[3];  const float* bk  = (const float*)d_in[4];
  const float* Wv  = (const float*)d_in[5];  const float* bv  = (const float*)d_in[6];
  const float* Wo  = (const float*)d_in[7];  const float* bo  = (const float*)d_in[8];
  const float* W1  = (const float*)d_in[9];  const float* b1  = (const float*)d_in[10];
  const float* W2  = (const float*)d_in[11]; const float* b2  = (const float*)d_in[12];
  const float* g1  = (const float*)d_in[13]; const float* be1 = (const float*)d_in[14];
  const float* g2  = (const float*)d_in[15]; const float* be2 = (const float*)d_in[16];

  char* ws = (char*)d_ws;
  size_t off = 0;
  auto alloc = [&](size_t bytes) -> void* {
    off = (off + 255) & ~(size_t)255;
    void* p = ws + off;
    off += bytes;
    return p;
  };

  bf16*  WtQ  = (bf16*)alloc((size_t)DM * DM * 2);
  bf16*  WtK  = (bf16*)alloc((size_t)DM * DM * 2);
  bf16*  WtV  = (bf16*)alloc((size_t)DM * DM * 2);
  bf16*  WtO  = (bf16*)alloc((size_t)DM * DM * 2);
  bf16*  Wt1  = (bf16*)alloc((size_t)DFF * DM * 2);   // [3072][768]
  bf16*  Wt2  = (bf16*)alloc((size_t)DM * DFF * 2);   // [768][3072]
  bf16*  Xbf  = (bf16*)alloc((size_t)NTOK * DM * 2);
  bf16*  Qb   = (bf16*)alloc((size_t)NBH * SEQL * HD * 2);
  bf16*  Kb   = (bf16*)alloc((size_t)NBH * SEQL * HD * 2);
  bf16*  Vtb  = (bf16*)alloc((size_t)NBH * HD * SEQL * 2);
  bf16*  AObf = (bf16*)alloc((size_t)NTOK * DM * 2);
  float* src2 = (float*)alloc((size_t)NTOK * DM * 4);
  float* xln  = (float*)alloc((size_t)NTOK * DM * 4);
  bf16*  xbf  = (bf16*)alloc((size_t)NTOK * DM * 2);
  bf16*  hbuf = (bf16*)alloc((size_t)NTOK * DFF * 2);
  float* ffo  = (float*)alloc((size_t)NTOK * DM * 4);

  const int nX = NTOK * DM;
  cvt_kernel<<<(nX + 255) / 256, 256, 0, stream>>>(src, Xbf, nX);
  tr_kernel<<<(DM*DM  + 255) / 256, 256, 0, stream>>>(Wq, WtQ, DM,  DM);
  tr_kernel<<<(DM*DM  + 255) / 256, 256, 0, stream>>>(Wk, WtK, DM,  DM);
  tr_kernel<<<(DM*DM  + 255) / 256, 256, 0, stream>>>(Wv, WtV, DM,  DM);
  tr_kernel<<<(DM*DM  + 255) / 256, 256, 0, stream>>>(Wo, WtO, DM,  DM);
  tr_kernel<<<(DM*DFF + 255) / 256, 256, 0, stream>>>(W1, Wt1, DM,  DFF);
  tr_kernel<<<(DFF*DM + 255) / 256, 256, 0, stream>>>(W2, Wt2, DFF, DM);

  dim3 gA(NTOK / 128, DM / 128);      // 64 x 6
  gemm_kernel<DM, DM, EPI_QK ><<<gA, 256, 0, stream>>>(Xbf, WtQ, bq, Qb);
  gemm_kernel<DM, DM, EPI_QK ><<<gA, 256, 0, stream>>>(Xbf, WtK, bk, Kb);
  gemm_kernel<DM, DM, EPI_V  ><<<gA, 256, 0, stream>>>(Xbf, WtV, bv, Vtb);

  attn_kernel<<<dim3(SEQL / 128, NBH), 256, 0, stream>>>(Qb, Kb, Vtb, AObf);

  gemm_kernel<DM, DM, EPI_F32><<<gA, 256, 0, stream>>>(AObf, WtO, bo, src2);
  ln_kernel<true><<<NTOK, 256, 0, stream>>>(src, src2, g1, be1, xln, xbf);

  dim3 gF1(NTOK / 128, DFF / 128);    // 64 x 24
  gemm_kernel<DM,  DFF, EPI_GELU><<<gF1, 256, 0, stream>>>(xbf, Wt1, b1, hbuf);
  gemm_kernel<DFF, DM,  EPI_F32 ><<<gA,  256, 0, stream>>>(hbuf, Wt2, b2, ffo);
  ln_kernel<false><<<NTOK, 256, 0, stream>>>(xln, ffo, g2, be2, (float*)d_out, nullptr);
}
